// GNN_based_VAE_37563783971071
// MI455X (gfx1250) — compile-verified, matrix operand order checked by reading, not run on hardware
//
#include <hip/hip_runtime.h>

typedef __attribute__((ext_vector_type(16))) _Float16 v16h;
typedef __attribute__((ext_vector_type(8)))  float    v8f;
typedef _Float16 h16;

#define S_ 2
#define B_ 16
#define N_ 64
#define F_ 4
#define D_ 128
#define E_ 4032
#define NROWS (S_*B_*N_)        // 2048 node rows
#define NS 132                  // padded LDS row stride (halves / floats)

// ---- workspace layout (bytes) ----
#define NODE_F32_OFFB 0u                         // 262144 f32
#define NODE_F16_OFFB (NODE_F32_OFFB + 262144u*4u)
#define ACC_OFFB      (NODE_F16_OFFB + 262144u*2u)
#define WF_OFFB       (ACC_OFFB      + 262144u*4u)
// fragment-linear f16 weights (offsets in halves, from WF base):
//   W1 region: [dir*2+k]*32768   (fwd1_0, fwd1_1, bwd1_0, bwd1_1)
//   W2 region: 131072 + [dir*2+k]*16384
//   e2n1: 196608   e2n2: 212992

// ---------------- WMMA helpers ----------------
__device__ __forceinline__ v8f wmma_f16(v16h a, v16h b, v8f c) {
  return __builtin_amdgcn_wmma_f32_16x16x32_f16(false, a, false, b, (short)0, c,
                                                false, false);
}

__device__ __forceinline__ v8f v8f_zero() {
  v8f z = {0.f, 0.f, 0.f, 0.f, 0.f, 0.f, 0.f, 0.f};
  return z;
}

// A-frag (16x32 f16) gathered from an LDS row-major buffer.
// ISA layout: lanes 0-15 hold M=lane, K={0..7,16..23}; lanes 16-31 K={8..15,24..31}.
__device__ __forceinline__ v16h load_a_frag(const h16* buf, int rowstride,
                                            int row, int k0, int g) {
  union { v16h v; unsigned u[8]; } r;
  const h16* p = buf + row * rowstride + k0 + 8 * g;
  #pragma unroll
  for (int v = 0; v < 8; ++v) {
    int kof = (v < 4) ? (2 * v) : (16 + 2 * (v - 4));
    r.u[v] = *(const unsigned*)(p + kof);
  }
  return r.v;
}

// B-frag from fragment-linear global f16 weights: contiguous 32B per lane.
__device__ __forceinline__ v16h load_b_frag(const h16* wf, int frag_idx, int lane) {
  return *(const v16h*)(wf + ((frag_idx * 32 + lane) << 4));
}

// ---------------- weight pre-swizzle (f32 KxN -> f16 fragment-linear) --------
__global__ void wprep_kernel(h16* __restrict__ dst, const float* __restrict__ src,
                             int KT) {
  int o = blockIdx.x * blockDim.x + threadIdx.x;
  if (o >= KT * 4096) return;                    // KT*8 frags * 512 halves
  int j = o & 15, lane = (o >> 4) & 31, fr = o >> 9;
  int ktile = fr % KT, ntile = fr / KT;
  int g = lane >> 4, v = j >> 1, c = j & 1;
  int kl = (v < 4) ? (2 * v + c + 8 * g) : (16 + 2 * (v - 4) + c + 8 * g);
  int k = ktile * 32 + kl;
  int n = ntile * 16 + (lane & 15);
  dst[o] = (h16)src[k * 128 + n];
}

// ---------------- encoder: x->h->emb, mean/logvar, reparameterize -----------
__global__ void encoder_kernel(const float* __restrict__ inp,
                               const int* __restrict__ feat_idx,
                               const float* __restrict__ W1, const float* __restrict__ b1,
                               const float* __restrict__ W2, const float* __restrict__ b2,
                               const float* __restrict__ eps,
                               float* __restrict__ out_mean, float* __restrict__ out_logvar,
                               float* __restrict__ node_f, h16* __restrict__ node_h) {
  __shared__ float xs[F_];
  __shared__ float hs[D_];
  int bn = blockIdx.x, b = bn >> 6, n = bn & 63;
  int j = threadIdx.x;                                    // 0..127
  if (j < F_) xs[j] = inp[b * (N_ * F_) + feat_idx[n * F_ + j]];
  __syncthreads();
  float h = b1[j];
  #pragma unroll
  for (int f = 0; f < F_; ++f) h += xs[f] * W1[f * D_ + j];
  hs[j] = h > 0.f ? h : 0.f;
  __syncthreads();
  float m = b2[j], lv = b2[D_ + j];
  for (int k = 0; k < D_; ++k) {
    float hv = hs[k];
    m  += hv * W2[k * (2 * D_) + j];
    lv += hv * W2[k * (2 * D_) + D_ + j];
  }
  lv = fminf(fmaxf(lv, -20.f), 20.f);
  out_mean  [b * (N_ * D_) + n * D_ + j] = m;
  out_logvar[b * (N_ * D_) + n * D_ + j] = lv;
  float sd = expf(0.5f * lv);
  #pragma unroll
  for (int s = 0; s < S_; ++s) {
    int row = (s * B_ + b) * N_ + n;
    float nv = m + sd * eps[row * D_ + j];
    node_f[row * D_ + j] = nv;
    node_h[row * D_ + j] = (h16)nv;
  }
}

__global__ void zero_kernel(float* __restrict__ p, int n) {
  int i = blockIdx.x * blockDim.x + threadIdx.x;
  if (i < n) p[i] = 0.f;
}

// ---------------- edge pass: fused gather + 4 MLPs + scatter-add ------------
// 4 waves/WG; each wave owns a 64-edge tile and uses a 4x4 WMMA register tile
// (M=64 x N=64 per pass) => 16 MACs/byte of LDS/L1 operand traffic.
#define EDGE_NW 4
#define TILE_M 64
#define EDGE_LDS (N_*NS*2 + EDGE_NW*TILE_M*NS*2 + N_*NS*4)   // 118272 B

__global__ __launch_bounds__(EDGE_NW * 32, 1)
void edge_kernel(const h16* __restrict__ node_h, const h16* __restrict__ wf,
                 const float* __restrict__ fwd1_b, const float* __restrict__ fwd2_b,
                 const float* __restrict__ bwd1_b, const float* __restrict__ bwd2_b,
                 const float* __restrict__ eweights, float* __restrict__ acc_g,
                 const int* __restrict__ only_forward) {
  extern __shared__ char smem[];
  h16*   node_sh = (h16*)smem;                           // 64 x NS halves
  h16*   hf_all  = node_sh + N_ * NS;                    // per-wave 64 x NS halves
  float* acc_sh  = (float*)(hf_all + EDGE_NW * TILE_M * NS);  // 64 x NS f32

  const int tid = threadIdx.x, lane = tid & 31, wave = tid >> 5;
  const int g = lane >> 4, lc = lane & 15;
  const int sb = blockIdx.x;                             // 0..31  (= s*B + b)
  const int s  = sb >> 4;
  const int fo = *only_forward;

  __builtin_prefetch(wf, 0, 1);
  __builtin_prefetch(wf + 131072, 0, 1);

  // stage this (s,b)'s node block into LDS; zero the accumulator
  const h16* src = node_h + sb * (N_ * D_);
  for (int i = tid; i < (N_ * D_) / 2; i += blockDim.x) {
    int row = i >> 6, cp = (i & 63) << 1;
    *(unsigned*)(node_sh + row * NS + cp) = ((const unsigned*)src)[i];
  }
  for (int i = tid; i < N_ * NS; i += blockDim.x) acc_sh[i] = 0.f;
  __syncthreads();

  const float* b1p[2] = {fwd1_b, bwd1_b};
  const float* b2p[2] = {fwd2_b, bwd2_b};
  const h16* w1base = wf;
  const h16* w2base = wf + 131072;
  const float* ew = eweights + s * (E_ * 2);
  h16* hf = hf_all + wave * TILE_M * NS;

  for (int t = blockIdx.y * EDGE_NW + wave; t < (E_ / TILE_M);
       t += gridDim.y * EDGE_NW) {
    const int e0 = t * TILE_M;
    // per-lane A-row indices for the 4 M-subtiles (row = m*16 + lane&15)
    int sndM[4], recM[4];
    #pragma unroll
    for (int m = 0; m < 4; ++m) {
      int eM = e0 + m * 16 + lc;
      int es = eM / 63, jj = eM - es * 63;
      sndM[m] = es;
      recM[m] = jj + (jj >= es ? 1 : 0);
    }

    const int ndir = fo ? 1 : 2;
    #pragma unroll 1
    for (int kt = 0; kt < 2; ++kt) {
      #pragma unroll 1
      for (int dir = 0; dir < ndir; ++dir) {
        const h16* w1 = w1base + (dir * 2 + kt) * 32768;
        const float* b1 = b1p[dir] + kt * 128;
        // ---- layer 1: ee(64x256) @ W1(256x128), bias+relu -> hf (f16) ----
        #pragma unroll
        for (int nh = 0; nh < 2; ++nh) {            // two 64-col halves
          v8f c[4][4];
          #pragma unroll
          for (int m = 0; m < 4; ++m)
            #pragma unroll
            for (int q = 0; q < 4; ++q) c[m][q] = v8f_zero();
          #pragma unroll
          for (int ktile = 0; ktile < 8; ++ktile) {
            v16h a[4], b[4];
            #pragma unroll
            for (int m = 0; m < 4; ++m)
              a[m] = load_a_frag(node_sh, NS, (ktile < 4) ? sndM[m] : recM[m],
                                 (ktile & 3) * 32, g);
            #pragma unroll
            for (int q = 0; q < 4; ++q)
              b[q] = load_b_frag(w1, (nh * 4 + q) * 8 + ktile, lane);
            #pragma unroll
            for (int m = 0; m < 4; ++m)
              #pragma unroll
              for (int q = 0; q < 4; ++q)
                c[m][q] = wmma_f16(a[m], b[q], c[m][q]);
          }
          float biasq[4];
          #pragma unroll
          for (int q = 0; q < 4; ++q) biasq[q] = b1[(nh * 4 + q) * 16 + lc];
          #pragma unroll
          for (int m = 0; m < 4; ++m)
            #pragma unroll
            for (int r = 0; r < 8; ++r) {
              int row = m * 16 + r + 8 * g;
              #pragma unroll
              for (int q = 0; q < 4; ++q) {
                float v = c[m][q][r] + biasq[q];
                v = v > 0.f ? v : 0.f;
                hf[row * NS + (nh * 4 + q) * 16 + lc] = (h16)v;
              }
            }
        }
        asm volatile("s_wait_dscnt 0x0" ::: "memory");
        __builtin_amdgcn_wave_barrier();
        // ---- layer 2: hf(64x128) @ W2(128x128), bias+relu, *w, scatter ----
        const h16* w2 = w2base + (dir * 2 + kt) * 16384;
        const float* b2 = b2p[dir] + kt * 128;
        #pragma unroll
        for (int nh = 0; nh < 2; ++nh) {
          v8f c[4][4];
          #pragma unroll
          for (int m = 0; m < 4; ++m)
            #pragma unroll
            for (int q = 0; q < 4; ++q) c[m][q] = v8f_zero();
          #pragma unroll
          for (int ktile = 0; ktile < 4; ++ktile) {
            v16h a[4], b[4];
            #pragma unroll
            for (int m = 0; m < 4; ++m)
              a[m] = load_a_frag(hf, NS, m * 16 + lc, ktile * 32, g);
            #pragma unroll
            for (int q = 0; q < 4; ++q)
              b[q] = load_b_frag(w2, (nh * 4 + q) * 4 + ktile, lane);
            #pragma unroll
            for (int m = 0; m < 4; ++m)
              #pragma unroll
              for (int q = 0; q < 4; ++q)
                c[m][q] = wmma_f16(a[m], b[q], c[m][q]);
          }
          float biasq[4];
          #pragma unroll
          for (int q = 0; q < 4; ++q) biasq[q] = b2[(nh * 4 + q) * 16 + lc];
          #pragma unroll
          for (int m = 0; m < 4; ++m)
            #pragma unroll
            for (int r = 0; r < 8; ++r) {
              int e = e0 + m * 16 + r + 8 * g;
              int es = e / 63, jj = e - es * 63;
              int rr = jj + (jj >= es ? 1 : 0);
              int tg = dir ? es : rr;
              float w = ew[e * 2 + kt];
              #pragma unroll
              for (int q = 0; q < 4; ++q) {
                float v = c[m][q][r] + biasq[q];
                v = v > 0.f ? v : 0.f;
                v *= w;
                __hip_atomic_fetch_add(&acc_sh[tg * NS + (nh * 4 + q) * 16 + lc],
                                       v, __ATOMIC_RELAXED,
                                       __HIP_MEMORY_SCOPE_WORKGROUP);
              }
            }
        }
        __builtin_amdgcn_wave_barrier();
      }
    }
  }
  __syncthreads();
  // flush LDS accumulator to global (other edge-chunk WGs add too)
  float* dst = acc_g + sb * (N_ * D_);
  for (int i = tid; i < N_ * D_; i += blockDim.x) {
    int row = i >> 7, col = i & 127;
    __hip_atomic_fetch_add(&dst[i], acc_sh[row * NS + col],
                           __ATOMIC_RELAXED, __HIP_MEMORY_SCOPE_AGENT);
  }
}

// ---------------- node pass: acc -> e2n1 -> e2n2, write node state ----------
#define NODE_NW 4
__global__ __launch_bounds__(NODE_NW * 32, 1)
void node_kernel(float* __restrict__ acc_g, const h16* __restrict__ wfe1,
                 const h16* __restrict__ wfe2, const float* __restrict__ e2n1_b,
                 const float* __restrict__ e2n2_b, float* __restrict__ node_f,
                 h16* __restrict__ node_h) {
  __shared__ h16 in_sh[NODE_NW * 16 * NS];
  __shared__ h16 hf_sh[NODE_NW * 16 * NS];
  const int tid = threadIdx.x, lane = tid & 31, wave = tid >> 5;
  const int g = lane >> 4, lc = lane & 15;
  const int R0 = (blockIdx.x * NODE_NW + wave) * 16;
  h16* in_w = in_sh + wave * 16 * NS;
  h16* hf_w = hf_sh + wave * 16 * NS;

  #pragma unroll
  for (int c = 0; c < 64; ++c) {
    int col = g * 64 + c;
    in_w[lc * NS + col] = (h16)acc_g[(R0 + lc) * D_ + col];
  }
  asm volatile("s_wait_dscnt 0x0" ::: "memory");
  __builtin_amdgcn_wave_barrier();

  v16h a[4];
  #pragma unroll
  for (int ktile = 0; ktile < 4; ++ktile)
    a[ktile] = load_a_frag(in_w, NS, lc, ktile * 32, g);
  #pragma unroll
  for (int nt = 0; nt < 8; ++nt) {
    v8f c = v8f_zero();
    #pragma unroll
    for (int ktile = 0; ktile < 4; ++ktile)
      c = wmma_f16(a[ktile], load_b_frag(wfe1, nt * 4 + ktile, lane), c);
    float bias = e2n1_b[nt * 16 + lc];
    #pragma unroll
    for (int r = 0; r < 8; ++r) {
      float v = c[r] + bias;
      hf_w[(r + 8 * g) * NS + nt * 16 + lc] = (h16)(v > 0.f ? v : 0.f);
    }
  }
  asm volatile("s_wait_dscnt 0x0" ::: "memory");
  __builtin_amdgcn_wave_barrier();

  #pragma unroll
  for (int ktile = 0; ktile < 4; ++ktile)
    a[ktile] = load_a_frag(hf_w, NS, lc, ktile * 32, g);
  #pragma unroll
  for (int nt = 0; nt < 8; ++nt) {
    v8f c = v8f_zero();
    #pragma unroll
    for (int ktile = 0; ktile < 4; ++ktile)
      c = wmma_f16(a[ktile], load_b_frag(wfe2, nt * 4 + ktile, lane), c);
    float bias = e2n2_b[nt * 16 + lc];
    #pragma unroll
    for (int r = 0; r < 8; ++r) {
      float v = c[r] + bias;
      v = v > 0.f ? v : 0.f;
      int row = R0 + r + 8 * g, col = nt * 16 + lc;
      node_f[row * D_ + col] = v;
      node_h[row * D_ + col] = (h16)v;
    }
  }
  __syncthreads();
  // zero accumulator rows for the next iteration (this block's 64 rows)
  float* z = acc_g + blockIdx.x * (NODE_NW * 16 * D_);
  for (int i = tid; i < NODE_NW * 16 * D_; i += blockDim.x) z[i] = 0.f;
}

// ---------------- final head: node -> fin1 -> fin2 -> out_final -------------
__global__ void final_kernel(const float* __restrict__ node_f,
                             const float* __restrict__ W1, const float* __restrict__ b1,
                             const float* __restrict__ W2, const float* __restrict__ b2,
                             float* __restrict__ out) {
  __shared__ float h1[D_];
  int row = blockIdx.x, j = threadIdx.x;          // 64 threads
  const float* nr = node_f + row * D_;
  #pragma unroll
  for (int rep = 0; rep < 2; ++rep) {
    int jj = j + rep * 64;
    float a = b1[jj];
    for (int k = 0; k < D_; ++k) a += nr[k] * W1[k * D_ + jj];
    h1[jj] = a > 0.f ? a : 0.f;
  }
  __syncthreads();
  if (j < F_) {
    float a = b2[j];
    for (int k = 0; k < D_; ++k) a += h1[k] * W2[k * F_ + j];
    out[row * F_ + j] = a;                        // (s,b,n*4+f) == row*4+f
  }
}

// ---------------- host launch ------------------------------------------------
extern "C" void kernel_launch(void* const* d_in, const int* in_sizes, int n_in,
                              void* d_out, int out_size, void* d_ws, size_t ws_size,
                              hipStream_t stream) {
  (void)in_sizes; (void)n_in; (void)out_size; (void)ws_size;
  const float* inp      = (const float*)d_in[0];
  const float* eweights = (const float*)d_in[1];
  const float* eps      = (const float*)d_in[4];
  const float* init1_w  = (const float*)d_in[5];
  const float* init1_b  = (const float*)d_in[6];
  const float* init2_w  = (const float*)d_in[7];
  const float* init2_b  = (const float*)d_in[8];
  const float* fwd1_w   = (const float*)d_in[9];
  const float* fwd1_b   = (const float*)d_in[10];
  const float* fwd2_w   = (const float*)d_in[11];
  const float* fwd2_b   = (const float*)d_in[12];
  const float* bwd1_w   = (const float*)d_in[13];
  const float* bwd1_b   = (const float*)d_in[14];
  const float* bwd2_w   = (const float*)d_in[15];
  const float* bwd2_b   = (const float*)d_in[16];
  const float* e2n1_w   = (const float*)d_in[17];
  const float* e2n1_b   = (const float*)d_in[18];
  const float* e2n2_w   = (const float*)d_in[19];
  const float* e2n2_b   = (const float*)d_in[20];
  const float* fin1_w   = (const float*)d_in[21];
  const float* fin1_b   = (const float*)d_in[22];
  const float* fin2_w   = (const float*)d_in[23];
  const float* fin2_b   = (const float*)d_in[24];
  const int*   feat_idx = (const int*)d_in[25];
  const int*   only_fwd = (const int*)d_in[26];

  float* out       = (float*)d_out;
  float* out_mean  = out + S_ * B_ * N_ * F_;                 // 8192
  float* out_lvar  = out_mean + B_ * N_ * D_;                 // +131072

  char* ws = (char*)d_ws;
  float* node_f = (float*)(ws + NODE_F32_OFFB);
  h16*   node_h = (h16*)(ws + NODE_F16_OFFB);
  float* acc    = (float*)(ws + ACC_OFFB);
  h16*   wf     = (h16*)(ws + WF_OFFB);

  hipFuncSetAttribute((const void*)edge_kernel,
                      hipFuncAttributeMaxDynamicSharedMemorySize, EDGE_LDS);

  // weight pre-swizzle (f32 -> f16 fragment-linear)
  wprep_kernel<<<128, 256, 0, stream>>>(wf + 0,      fwd1_w + 0,     8);
  wprep_kernel<<<128, 256, 0, stream>>>(wf + 32768,  fwd1_w + 32768, 8);
  wprep_kernel<<<128, 256, 0, stream>>>(wf + 65536,  bwd1_w + 0,     8);
  wprep_kernel<<<128, 256, 0, stream>>>(wf + 98304,  bwd1_w + 32768, 8);
  wprep_kernel<<<64, 256, 0, stream>>>(wf + 131072,           fwd2_w + 0,     4);
  wprep_kernel<<<64, 256, 0, stream>>>(wf + 131072 + 16384,   fwd2_w + 16384, 4);
  wprep_kernel<<<64, 256, 0, stream>>>(wf + 131072 + 32768,   bwd2_w + 0,     4);
  wprep_kernel<<<64, 256, 0, stream>>>(wf + 131072 + 49152,   bwd2_w + 16384, 4);
  wprep_kernel<<<64, 256, 0, stream>>>(wf + 196608, e2n1_w, 4);
  wprep_kernel<<<64, 256, 0, stream>>>(wf + 212992, e2n2_w, 4);

  encoder_kernel<<<B_ * N_, D_, 0, stream>>>(inp, feat_idx, init1_w, init1_b,
                                             init2_w, init2_b, eps, out_mean,
                                             out_lvar, node_f, node_h);
  zero_kernel<<<(NROWS * D_ + 255) / 256, 256, 0, stream>>>(acc, NROWS * D_);

  for (int it = 0; it < 3; ++it) {
    edge_kernel<<<dim3(S_ * B_, 8), EDGE_NW * 32, EDGE_LDS, stream>>>(
        node_h, wf, fwd1_b, fwd2_b, bwd1_b, bwd2_b, eweights, acc, only_fwd);
    node_kernel<<<NROWS / (NODE_NW * 16), NODE_NW * 32, 0, stream>>>(
        acc, wf + 196608, wf + 212992, e2n1_b, e2n2_b, node_f, node_h);
  }
  final_kernel<<<NROWS, 64, 0, stream>>>(node_f, fin1_w, fin1_b, fin2_w, fin2_b, out);
}
